// MixtureOfExperts_64888365908350
// MI455X (gfx1250) — matrix operand order repaired and checked by reading, not compiled, not run on hardware
//
#include <hip/hip_runtime.h>
#include <hip/hip_bf16.h>

typedef __attribute__((ext_vector_type(16))) __bf16        v16bf;
typedef __attribute__((ext_vector_type(8)))  float         v8f;
typedef __attribute__((ext_vector_type(8)))  unsigned int  v8u;

// ---------------- problem constants ----------------
#define DMODEL   1024
#define DFF      4096
#define NEXP     8
#define TTOK     8192            // 4*2048 tokens
#define NPAIR    (TTOK*2)        // top-2
#define CAPROWS  (NPAIR + NEXP*16)   // each expert segment padded to 16
#define NTILES   (CAPROWS/16)

#define OUT_TOK_ELEMS ((size_t)TTOK*DMODEL)  // 8,388,608 ; aux at [OUT_TOK_ELEMS]

// fragments: one B fragment = 32x16 bf16, stored lane-major: 32 lanes * 16 bf16 = 512 elems
#define F1_FRAGS (NEXP*(DFF/16)*(DMODEL/32))   // 8*256*32 = 65536
#define F2_FRAGS (NEXP*(DMODEL/16)*(DFF/32))   // 8*64*128 = 65536

#define H_STRIDE (DFF + 8)                     // LDS padding vs bank conflicts
#define SMEM_BYTES (128 + 16*H_STRIDE*2)       // tok[16],w[16] + H tile

// ---------------- workspace layout (bytes) ----------------
#define W1S_OFF   ((size_t)0)
#define W1S_BYTES ((size_t)F1_FRAGS*512*2)
#define W2S_OFF   (W1S_OFF + W1S_BYTES)
#define W2S_BYTES ((size_t)F2_FRAGS*512*2)
#define XG_OFF    (W2S_OFF + W2S_BYTES)
#define XG_ELEMS  ((size_t)CAPROWS*DMODEL)
#define XG_BYTES  (XG_ELEMS*2)
#define PTOK_OFF  (XG_OFF + XG_BYTES)
#define PW_OFF    (PTOK_OFF + (size_t)CAPROWS*4)
#define SEL_OFF   (PW_OFF  + (size_t)CAPROWS*4)
#define SELW_OFF  (SEL_OFF + (size_t)NPAIR*4)
#define CNT_OFF   (SELW_OFF + (size_t)NPAIR*4)     // float[8] counts
#define PSUM_OFF  (CNT_OFF + 32)                   // float[8] prob sums
#define CUR_OFF   (PSUM_OFF + 32)                  // int[8] cursors
#define OFFS_OFF  (CUR_OFF + 32)                   // int[9] segment offsets
#define TOTR_OFF  (OFFS_OFF + 40)                  // int total padded rows

__device__ __forceinline__ unsigned short f2bf(float x) {
    unsigned int u = __builtin_bit_cast(unsigned int, x);
    u += 0x7FFFu + ((u >> 16) & 1u);          // round-to-nearest-even
    return (unsigned short)(u >> 16);
}

__device__ __forceinline__ v16bf load_a_frag(const unsigned short* p) {
    uint4 c0 = *(const uint4*)(p);
    uint4 c1 = *(const uint4*)(p + 8);
    v8u u; u[0]=c0.x; u[1]=c0.y; u[2]=c0.z; u[3]=c0.w;
           u[4]=c1.x; u[5]=c1.y; u[6]=c1.z; u[7]=c1.w;
    return __builtin_bit_cast(v16bf, u);
}

// ---------------- 0: zero helper ----------------
__global__ void moe_zero_f32(float* __restrict__ p, size_t n) {
    size_t i = (size_t)blockIdx.x * blockDim.x + threadIdx.x;
    if (i < n) p[i] = 0.f;
}

// ---------------- 1: init pair arrays + gathered-A region ----------------
__global__ void moe_init_pairs(unsigned int* __restrict__ xg_u32,
                               int* __restrict__ ptok, float* __restrict__ pw) {
    size_t stride = (size_t)gridDim.x * blockDim.x;
    size_t gid = (size_t)blockIdx.x * blockDim.x + threadIdx.x;
    size_t nXg = XG_ELEMS / 2;
    for (size_t i = gid; i < nXg; i += stride) xg_u32[i] = 0u;
    for (size_t i = gid; i < CAPROWS; i += stride) { ptok[i] = -1; pw[i] = 0.f; }
}

// ---------------- 2: weight fp32 -> bf16, swizzled to WMMA B-fragment layout ----
// B frag (32x16): lane l holds column n = n0+(l&15), K = k0 + ((l&16)?16:0) + j, j=0..15
__global__ void moe_convert_w(const float* __restrict__ w1, const float* __restrict__ w2,
                              unsigned short* __restrict__ w1s, unsigned short* __restrict__ w2s) {
    int wg = blockIdx.x * 8 + (threadIdx.x >> 5);
    int lane = threadIdx.x & 31;
    const float* src; unsigned short* dst; int Ndim, kbase, n;
    if (wg < F1_FRAGS) {
        int e = wg >> 13, rem = wg & 8191, nt = rem >> 5, kc = rem & 31;
        Ndim = DFF; n = nt * 16 + (lane & 15);
        kbase = kc * 32 + ((lane & 16) ? 16 : 0);
        src = w1 + (size_t)e * DMODEL * DFF;
        dst = w1s + (size_t)wg * 512 + lane * 16;
    } else {
        int g = wg - F1_FRAGS;
        int e = g >> 13, rem = g & 8191, nt = rem >> 7, kc = rem & 127;
        Ndim = DMODEL; n = nt * 16 + (lane & 15);
        kbase = kc * 32 + ((lane & 16) ? 16 : 0);
        src = w2 + (size_t)e * DFF * DMODEL;
        dst = w2s + (size_t)g * 512 + lane * 16;
    }
    unsigned int packed[8];
#pragma unroll
    for (int w = 0; w < 8; ++w) {
        float a = src[(size_t)(kbase + 2 * w) * Ndim + n];
        float b = src[(size_t)(kbase + 2 * w + 1) * Ndim + n];
        packed[w] = (unsigned int)f2bf(a) | ((unsigned int)f2bf(b) << 16);
    }
    *(uint4*)dst       = make_uint4(packed[0], packed[1], packed[2], packed[3]);
    *(uint4*)(dst + 8) = make_uint4(packed[4], packed[5], packed[6], packed[7]);
}

// ---------------- 3: gating (one wave per token) ----------------
__global__ void moe_gate(const float* __restrict__ x, const float* __restrict__ gw,
                         const float* __restrict__ gb,
                         int* __restrict__ sel, float* __restrict__ selw,
                         float* __restrict__ counts, float* __restrict__ psum) {
    int token = blockIdx.x * 8 + (threadIdx.x >> 5);
    int lane = threadIdx.x & 31;
    float acc[NEXP];
#pragma unroll
    for (int e = 0; e < NEXP; ++e) acc[e] = 0.f;
    const float* xr = x + (size_t)token * DMODEL;
    for (int d = lane; d < DMODEL; d += 32) {
        float xv = xr[d];
        const float* g = gw + (size_t)d * NEXP;
#pragma unroll
        for (int e = 0; e < NEXP; ++e) acc[e] += xv * g[e];
    }
#pragma unroll
    for (int e = 0; e < NEXP; ++e)
        for (int off = 16; off > 0; off >>= 1) acc[e] += __shfl_down(acc[e], off);
    if (lane == 0) {
        float lg[NEXP];
#pragma unroll
        for (int e = 0; e < NEXP; ++e) lg[e] = acc[e] + gb[e];
        int i0 = 0;
        for (int e = 1; e < NEXP; ++e) if (lg[e] > lg[i0]) i0 = e;
        int i1 = (i0 == 0) ? 1 : 0;
        for (int e = 0; e < NEXP; ++e) if (e != i0 && lg[e] > lg[i1]) i1 = e;
        float e1 = __expf(lg[i1] - lg[i0]);
        float w0 = 1.f / (1.f + e1), w1 = e1 / (1.f + e1);
        // full softmax for aux loss (i0 is argmax)
        float s = 0.f, p[NEXP];
#pragma unroll
        for (int e = 0; e < NEXP; ++e) { p[e] = __expf(lg[e] - lg[i0]); s += p[e]; }
        float inv = 1.f / s;
#pragma unroll
        for (int e = 0; e < NEXP; ++e) atomicAdd(&psum[e], p[e] * inv);
        atomicAdd(&counts[i0], 1.f);
        atomicAdd(&counts[i1], 1.f);
        sel[2 * token]     = i0; selw[2 * token]     = w0;
        sel[2 * token + 1] = i1; selw[2 * token + 1] = w1;
    }
}

// ---------------- 4: scan (segment offsets, cursors, aux loss) ----------------
__global__ void moe_scan(const float* __restrict__ counts, const float* __restrict__ psum,
                         int* __restrict__ offs, int* __restrict__ cur,
                         int* __restrict__ totr, float* __restrict__ out_aux) {
    if (threadIdx.x == 0) {
        int off = 0;
        for (int e = 0; e < NEXP; ++e) {
            offs[e] = off; cur[e] = off;
            int c = (int)counts[e];
            off += (c + 15) & ~15;
        }
        offs[NEXP] = off; *totr = off;
        float aux = 0.f;
        for (int e = 0; e < NEXP; ++e)
            aux += (counts[e] / (float)(TTOK * 2)) * (psum[e] / (float)TTOK);
        *out_aux = (float)NEXP * aux;
    }
}

// ---------------- 5: scatter tokens -> bf16 expert-segmented A rows ------------
__global__ void moe_scatter(const float* __restrict__ x,
                            const int* __restrict__ sel, const float* __restrict__ selw,
                            int* __restrict__ cur, int* __restrict__ ptok,
                            float* __restrict__ pw, unsigned short* __restrict__ xg) {
    int p = blockIdx.x * 8 + (threadIdx.x >> 5);
    int lane = threadIdx.x & 31;
    int token = p >> 1;
    int e = sel[p];
    float w = selw[p];
    int pos = 0;
    if (lane == 0) pos = atomicAdd(&cur[e], 1);
    pos = __shfl(pos, 0);
    if (lane == 0) { ptok[pos] = token; pw[pos] = w; }
    const float* xr = x + (size_t)token * DMODEL;
    unsigned short* dr = xg + (size_t)pos * DMODEL;
#pragma unroll
    for (int j = 0; j < DMODEL / 32; ++j) {
        int d = j * 32 + lane;
        dr[d] = f2bf(xr[d]);
    }
}

// ---------------- 6: fused MoE FFN tile kernel (16 tokens / block, 8 waves) ----
// Register blocking: 8 accumulator tiles per wave; kc outer loop loads one A
// fragment and feeds 8 WMMAs -> no compiler hoisting/spilling of fragments.
__global__ void moe_gemm(const unsigned short* __restrict__ xg,
                         const unsigned short* __restrict__ w1s,
                         const unsigned short* __restrict__ w2s,
                         const float* __restrict__ b1, const float* __restrict__ b2,
                         const int* __restrict__ ptok, const float* __restrict__ pw,
                         const int* __restrict__ offs, const int* __restrict__ totr,
                         float* __restrict__ out) {
    extern __shared__ char smem[];
    int*            tok_s = (int*)smem;
    float*          w_s   = (float*)(smem + 64);
    unsigned short* Hs    = (unsigned short*)(smem + 128);

    int row0 = blockIdx.x * 16;
    if (row0 >= *totr) return;                 // uniform exit
    int e = 0;
    while (e < NEXP - 1 && row0 >= offs[e + 1]) ++e;

    int wave = threadIdx.x >> 5;
    int lane = threadIdx.x & 31;
    if (threadIdx.x < 16) { tok_s[threadIdx.x] = ptok[row0 + threadIdx.x];
                            w_s[threadIdx.x]   = pw[row0 + threadIdx.x]; }

    int m   = lane & 15;
    int hi8 = (lane & 16) ? 16 : 0;
    int mlo = (lane & 16) ? 8 : 0;
    const unsigned short* Arow = xg + (size_t)(row0 + m) * DMODEL + hi8;

    // ---- GEMM1: H = gelu(X @ W1 + b1); wave owns 512 cols = 4 groups x 8 tiles
    for (int g = 0; g < 4; ++g) {
        int nt0 = wave * 32 + g * 8;
        v8f acc[8];
#pragma unroll
        for (int i = 0; i < 8; ++i) acc[i] = (v8f){0.f,0.f,0.f,0.f,0.f,0.f,0.f,0.f};
        // B fragments for tile nt0+i at: gbase + i*(32*512) + kc*512
        const unsigned short* gbase =
            w1s + (((size_t)e * 256 + nt0) * 32) * 512 + lane * 16;
#pragma unroll 1
        for (int kc = 0; kc < 32; ++kc) {
            v16bf a = load_a_frag(Arow + kc * 32);
#pragma unroll
            for (int i = 0; i < 8; ++i) {
                v8u ub = *(const v8u*)(gbase + (size_t)i * 16384 + (size_t)kc * 512);
                v16bf b = __builtin_bit_cast(v16bf, ub);
                acc[i] = __builtin_amdgcn_wmma_f32_16x16x32_bf16(
                             false, a, false, b, (short)0, acc[i], false, false);
            }
        }
#pragma unroll
        for (int i = 0; i < 8; ++i) {
            int n = (nt0 + i) * 16 + (lane & 15);
            float bias = b1[(size_t)e * DFF + n];
#pragma unroll
            for (int r = 0; r < 8; ++r) {
                float h = acc[i][r] + bias;
                float gv = 0.5f * h * (1.f + erff(h * 0.70710678118654752f));
                Hs[(r + mlo) * H_STRIDE + n] = f2bf(gv);
            }
        }
    }
    __syncthreads();

    // ---- GEMM2: out += w * (H @ W2 + b2); wave owns 128 cols = 8 tiles ----
    {
        const unsigned short* Hrow = Hs + (size_t)m * H_STRIDE + hi8;
        int nt0 = wave * 8;
        v8f acc[8];
#pragma unroll
        for (int i = 0; i < 8; ++i) acc[i] = (v8f){0.f,0.f,0.f,0.f,0.f,0.f,0.f,0.f};
        // B fragments for tile nt0+i at: gbase + i*(128*512) + kc*512
        const unsigned short* gbase =
            w2s + (((size_t)e * 64 + nt0) * 128) * 512 + lane * 16;
#pragma unroll 1
        for (int kc = 0; kc < 128; ++kc) {
            v16bf a = load_a_frag(Hrow + kc * 32);   // ds_load_b128 x2
#pragma unroll
            for (int i = 0; i < 8; ++i) {
                v8u ub = *(const v8u*)(gbase + (size_t)i * 65536 + (size_t)kc * 512);
                v16bf b = __builtin_bit_cast(v16bf, ub);
                acc[i] = __builtin_amdgcn_wmma_f32_16x16x32_bf16(
                             false, a, false, b, (short)0, acc[i], false, false);
            }
        }
#pragma unroll
        for (int i = 0; i < 8; ++i) {
            int n = (nt0 + i) * 16 + (lane & 15);
            float bias = b2[(size_t)e * DMODEL + n];
#pragma unroll
            for (int r = 0; r < 8; ++r) {
                int mm = r + mlo;
                int t = tok_s[mm];
                float val = w_s[mm] * (acc[i][r] + bias);
                if (t >= 0) atomicAdd(&out[(size_t)t * DMODEL + n], val);
            }
        }
    }
}

// ---------------- host launcher ----------------
extern "C" void kernel_launch(void* const* d_in, const int* in_sizes, int n_in,
                              void* d_out, int out_size, void* d_ws, size_t ws_size,
                              hipStream_t stream) {
    const float* x  = (const float*)d_in[0];
    const float* gw = (const float*)d_in[1];
    const float* gb = (const float*)d_in[2];
    const float* w1 = (const float*)d_in[3];
    const float* b1 = (const float*)d_in[4];
    const float* w2 = (const float*)d_in[5];
    const float* b2 = (const float*)d_in[6];
    float* out = (float*)d_out;

    char* ws = (char*)d_ws;
    unsigned short* w1s  = (unsigned short*)(ws + W1S_OFF);
    unsigned short* w2s  = (unsigned short*)(ws + W2S_OFF);
    unsigned short* xg   = (unsigned short*)(ws + XG_OFF);
    int*            ptok = (int*)(ws + PTOK_OFF);
    float*          pw   = (float*)(ws + PW_OFF);
    int*            sel  = (int*)(ws + SEL_OFF);
    float*          selw = (float*)(ws + SELW_OFF);
    float*          cnt  = (float*)(ws + CNT_OFF);
    float*          psum = (float*)(ws + PSUM_OFF);
    int*            cur  = (int*)(ws + CUR_OFF);
    int*            offs = (int*)(ws + OFFS_OFF);
    int*            totr = (int*)(ws + TOTR_OFF);

    (void)in_sizes; (void)n_in; (void)out_size; (void)ws_size;

    hipFuncSetAttribute((const void*)moe_gemm,
                        hipFuncAttributeMaxDynamicSharedMemorySize, SMEM_BYTES);

    size_t outN = OUT_TOK_ELEMS + 1;
    moe_zero_f32<<<(unsigned)((outN + 255) / 256), 256, 0, stream>>>(out, outN);
    moe_zero_f32<<<1, 256, 0, stream>>>(cnt, 16);   // counts[8] + psum[8] contiguous
    moe_init_pairs<<<4096, 256, 0, stream>>>((unsigned int*)xg, ptok, pw);
    moe_convert_w<<<(F1_FRAGS + F2_FRAGS) / 8, 256, 0, stream>>>(w1, w2, w1s, w2s);
    moe_gate<<<TTOK / 8, 256, 0, stream>>>(x, gw, gb, sel, selw, cnt, psum);
    moe_scan<<<1, 32, 0, stream>>>(cnt, psum, offs, cur, totr, out + OUT_TOK_ELEMS);
    moe_scatter<<<NPAIR / 8, 256, 0, stream>>>(x, sel, selw, cur, ptok, pw, xg);
    moe_gemm<<<NTILES, 256, SMEM_BYTES, stream>>>(xg, w1s, w2s, b1, b2,
                                                  ptok, pw, offs, totr, out);
}
